// StandardTopKSAE_16939351015444
// MI455X (gfx1250) — compile-verified, matrix operand order checked by reading, not run on hardware
//
#include <hip/hip_runtime.h>

// ---------------------------------------------------------------------------
// CDNA5 (gfx1250) TopK-SAE forward.
//   K1: encoder GEMM  (bf16 WMMA, fp32 accum)  pre_acts -> acts region of d_out
//   K2: per-row exact top-k threshold (TDM tensor_load_to_lds + radix select)
//       + relu*mask in place
//   K3: decoder GEMM  (same WMMA kernel)       acts @ W_dec^T -> recon region
// ---------------------------------------------------------------------------

typedef __attribute__((ext_vector_type(16))) __bf16 v16bf;
typedef __attribute__((ext_vector_type(8)))  __bf16 v8bf;
typedef __attribute__((ext_vector_type(4)))  __bf16 v4bf;
typedef __attribute__((ext_vector_type(8)))  float  v8f;
typedef __attribute__((ext_vector_type(4)))  unsigned int v4u;
typedef __attribute__((ext_vector_type(8)))  int    v8i;
typedef __attribute__((ext_vector_type(4)))  int    v4i;

#define BM 128
#define BN 128
#define BK 32
#define LDSS 48   // bf16 elems per LDS row: 96B stride (16B multiple, bank-skewed)

// C[M,N] = A[M,K] * B[N,K]^T  (+bias[N] if bias != nullptr)
// Both operands K-major ("NT" GEMM) -> identical fragment addressing for A & B.
__global__ __launch_bounds__(256)
void sae_gemm_bf16_nt(const float* __restrict__ A, const float* __restrict__ B,
                      const float* __restrict__ bias, float* __restrict__ C,
                      int K, int lda, int ldb, int ldc)
{
    __shared__ __align__(16) __bf16 As[2][BM][LDSS];
    __shared__ __align__(16) __bf16 Bs[2][BN][LDSS];

    const int tid  = threadIdx.x;
    const int lane = tid & 31;          // wave32
    const int wave = tid >> 5;          // 8 waves
    const int wm   = wave >> 2;         // 0..1  (M dir, 64 rows each)
    const int wn   = wave & 3;          // 0..3  (N dir, 32 cols each)
    const int half = lane >> 4;         // K-half select per ISA A/B layout
    const int l16  = lane & 15;

    const int rowBlock = blockIdx.y * BM;
    const int colBlock = blockIdx.x * BN;

    // Each thread stages 4 float4 of A and 4 of B per K-step (128x32 fp32 tile).
    float4 aReg[4], bReg[4];

    auto loadTiles = [&](int k0) {
#pragma unroll
        for (int i = 0; i < 4; ++i) {
            int q = tid + i * 256;      // 0..1023
            int r = q >> 3, c4 = q & 7; // row 0..127, float4-col 0..7
            aReg[i] = *(const float4*)(A + (size_t)(rowBlock + r) * lda + k0 + c4 * 4);
            bReg[i] = *(const float4*)(B + (size_t)(colBlock + r) * ldb + k0 + c4 * 4);
        }
    };
    auto stashTiles = [&](int buf) {
#pragma unroll
        for (int i = 0; i < 4; ++i) {
            int q = tid + i * 256;
            int r = q >> 3, c4 = q & 7;
            v4bf a4 = { (__bf16)aReg[i].x, (__bf16)aReg[i].y,
                        (__bf16)aReg[i].z, (__bf16)aReg[i].w };
            v4bf b4 = { (__bf16)bReg[i].x, (__bf16)bReg[i].y,
                        (__bf16)bReg[i].z, (__bf16)bReg[i].w };
            *(v4bf*)&As[buf][r][c4 * 4] = a4;   // 8B ds_store
            *(v4bf*)&Bs[buf][r][c4 * 4] = b4;
        }
    };

    v8f acc[4][2];
#pragma unroll
    for (int mt = 0; mt < 4; ++mt)
#pragma unroll
        for (int nt = 0; nt < 2; ++nt)
#pragma unroll
            for (int e = 0; e < 8; ++e) acc[mt][nt][e] = 0.0f;

    loadTiles(0);
    stashTiles(0);
    __syncthreads();

    const int nK = K / BK;
    const int kb = half * 8;    // lanes 0-15: K {0-7,16-23}; lanes 16-31: {8-15,24-31}

    for (int kt = 0; kt < nK; ++kt) {
        const int buf = kt & 1;
        if (kt + 1 < nK) loadTiles((kt + 1) * BK);   // prefetch next tile to VGPRs

        // ---- B fragments (2x 32x16, bf16) ----
        v16bf bfrag[2];
#pragma unroll
        for (int nt = 0; nt < 2; ++nt) {
            const __bf16* bp = &Bs[buf][wn * 32 + nt * 16 + l16][kb];
            v8bf lo = *(const v8bf*)bp;          // ds_load_b128
            v8bf hi = *(const v8bf*)(bp + 16);   // ds_load_b128
#pragma unroll
            for (int e = 0; e < 8; ++e) { bfrag[nt][e] = lo[e]; bfrag[nt][e + 8] = hi[e]; }
        }
        // ---- A fragments + WMMA (4x2 tiles of 16x16) ----
#pragma unroll
        for (int mt = 0; mt < 4; ++mt) {
            const __bf16* ap = &As[buf][wm * 64 + mt * 16 + l16][kb];
            v8bf lo = *(const v8bf*)ap;
            v8bf hi = *(const v8bf*)(ap + 16);
            v16bf afrag;
#pragma unroll
            for (int e = 0; e < 8; ++e) { afrag[e] = lo[e]; afrag[e + 8] = hi[e]; }
#pragma unroll
            for (int nt = 0; nt < 2; ++nt) {
                acc[mt][nt] = __builtin_amdgcn_wmma_f32_16x16x32_bf16(
                    false, afrag, false, bfrag[nt], (short)0, acc[mt][nt], false, false);
            }
        }

        if (kt + 1 < nK) stashTiles(buf ^ 1);    // fill other buffer
        __syncthreads();
    }

    // Epilogue: C/D layout — VGPR r holds M=r (lanes 0-15) / M=r+8 (lanes 16-31).
#pragma unroll
    for (int nt = 0; nt < 2; ++nt) {
        const int col = colBlock + wn * 32 + nt * 16 + l16;
        const float bv = bias ? bias[col] : 0.0f;
#pragma unroll
        for (int mt = 0; mt < 4; ++mt)
#pragma unroll
            for (int r = 0; r < 8; ++r) {
                const int row = rowBlock + wm * 64 + mt * 16 + half * 8 + r;
                C[(size_t)row * ldc + col] = acc[mt][nt][r] + bv;
            }
    }
}

// ---------------------------------------------------------------------------
// K2: one workgroup per row. Stage the whole 24576-float row into LDS with a
// single TDM tensor_load_to_lds (TENSORcnt-tracked, 96 KB of the 320 KB WGP
// LDS), transform in place to order-preserving uints, bitwise radix-select the
// exact K-th largest, then write back relu(pre)*mask in place.
// ---------------------------------------------------------------------------
__global__ __launch_bounds__(256)
void topk_mask_relu(float* __restrict__ acts, int cols, const int* __restrict__ kptr)
{
    extern __shared__ unsigned int u[];   // cols entries (raw bits, then monotone)
    __shared__ unsigned int cnt;

    const int row = blockIdx.x;
    const int tid = threadIdx.x;
    float* rp = acts + (size_t)row * cols;
    const unsigned int k = (unsigned int)(*kptr);

#if __has_builtin(__builtin_amdgcn_tensor_load_to_lds) && \
    __has_builtin(__builtin_amdgcn_s_wait_tensorcnt)
    // ---- TDM bulk copy: global row -> LDS (one instruction, one wave) ----
    if (tid < 32) {
        const unsigned long long gaddr = (unsigned long long)(const void*)rp;
        // generic LDS pointer: low 32 bits == byte offset in wave's LDS space
        const unsigned int lds_off = (unsigned int)(unsigned long long)(const void*)u;
        const unsigned int dim0 = (unsigned int)cols;     // elements (fp32)

        v4u g0;
        g0[0] = 1u;                                   // count=1, user mode, no gather
        g0[1] = lds_off;                              // D#.lds_addr (bytes)
        g0[2] = (unsigned int)(gaddr & 0xffffffffull);        // global_addr[31:0]
        g0[3] = (unsigned int)((gaddr >> 32) & 0x1ffffffull)  // global_addr[56:32]
                | 0x80000000u;                                // type=2 ("image")

        v8i g1;
        g1[0] = (int)(2u << 16);          // workgroup_mask=0, data_size=2 (4B), no pad
        g1[1] = (int)((dim0 & 0xffffu) << 16);            // tensor_dim0[15:0] -> [31:16]
        g1[2] = (int)(((dim0 >> 16) & 0xffffu))           // tensor_dim0[31:16]
                | (int)(1u << 16);                        // tensor_dim1 = 1
        g1[3] = (int)((dim0 & 0xffffu) << 16);            // tile_dim0 = cols (<=65535)
        g1[4] = (int)1;                                   // tile_dim1 = 1, tile_dim2 = 0
        g1[5] = (int)dim0;                                // tensor_dim0_stride[31:0]
        g1[6] = 0;                                        // stride hi / dim1_stride lo
        g1[7] = 0;

        v4i gz4 = {0, 0, 0, 0};                           // groups 2/3 unused (2D)
        v8i gz8 = {0, 0, 0, 0, 0, 0, 0, 0};
        // 6-arg (clang-23 / therock-10.0) form of the TDM builtin.
        __builtin_amdgcn_tensor_load_to_lds(g0, g1, gz4, gz4, gz8, 0 /*cpol*/);
        __builtin_amdgcn_s_wait_tensorcnt(0);
    }
    __syncthreads();
    // float bits -> monotone uint (ascending), in place in LDS
    for (int i = tid; i < cols; i += 256) {
        unsigned int b = u[i];
        u[i] = (b & 0x80000000u) ? ~b : (b | 0x80000000u);
    }
#else
    // Fallback: cooperative vector loads
    for (int i = tid; i < cols; i += 256) {
        unsigned int b = __float_as_uint(rp[i]);
        u[i] = (b & 0x80000000u) ? ~b : (b | 0x80000000u);
    }
#endif
    __syncthreads();

    // Largest T with count(u >= T) >= k  ==  K-th largest value.
    unsigned int prefix = 0;
    for (int bit = 31; bit >= 0; --bit) {
        const unsigned int cand = prefix | (1u << bit);
        if (tid == 0) cnt = 0;
        __syncthreads();
        unsigned int local = 0;
        for (int i = tid; i < cols; i += 256) local += (u[i] >= cand) ? 1u : 0u;
#pragma unroll
        for (int off = 16; off > 0; off >>= 1) local += __shfl_down(local, off, 32);
        if ((tid & 31) == 0) atomicAdd(&cnt, local);
        __syncthreads();
        if (cnt >= k) prefix = cand;
        __syncthreads();
    }

    // acts = (in top-k) ? relu(pre) : 0
    for (int i = tid; i < cols; i += 256) {
        const unsigned int uv = u[i];
        const unsigned int b  = (uv & 0x80000000u) ? (uv & 0x7fffffffu) : ~uv;
        const float f = __uint_as_float(b);
        rp[i] = (uv >= prefix) ? fmaxf(f, 0.0f) : 0.0f;
    }
}

// ---------------------------------------------------------------------------
extern "C" void kernel_launch(void* const* d_in, const int* in_sizes, int n_in,
                              void* d_out, int out_size, void* d_ws, size_t ws_size,
                              hipStream_t stream)
{
    (void)in_sizes; (void)n_in; (void)out_size; (void)d_ws; (void)ws_size;

    const float* x     = (const float*)d_in[0];   // [4096, 768]
    const float* W_enc = (const float*)d_in[1];   // [24576, 768]
    const float* b_enc = (const float*)d_in[2];   // [24576]
    const float* W_dec = (const float*)d_in[3];   // [768, 24576]
    const int*   kptr  = (const int*)d_in[4];     // scalar k

    const int B = 4096, D = 768, F = 24576;
    float* recon = (float*)d_out;                       // [4096, 768]
    float* acts  = (float*)d_out + (size_t)B * D;       // [4096, 24576]

    dim3 blk(256);

    // 1) encoder: pre_acts = x @ W_enc^T + b_enc  -> acts region
    dim3 g1(F / BN, B / BM);
    sae_gemm_bf16_nt<<<g1, blk, 0, stream>>>(x, W_enc, b_enc, acts,
                                             /*K=*/D, /*lda=*/D, /*ldb=*/D, /*ldc=*/F);

    // 2) top-k threshold + relu*mask, in place (96 KB dynamic LDS per WG)
    topk_mask_relu<<<dim3(B), blk, F * sizeof(unsigned int), stream>>>(acts, F, kptr);

    // 3) decoder: recon = acts @ W_dec^T (W_dec is [N=768 rows, K=24576] row-major)
    dim3 g3(D / BN, B / BM);
    sae_gemm_bf16_nt<<<g3, blk, 0, stream>>>(acts, W_dec, nullptr, recon,
                                             /*K=*/F, /*lda=*/F, /*ldb=*/F, /*ldc=*/D);
}
